// IterativeStructuralRefinement_89129161326848
// MI455X (gfx1250) — compile-verified, matrix operand order checked by reading, not run on hardware
//
#include <hip/hip_runtime.h>
#include <math.h>

#define LAYERS 12
#define BATCH  8
#define SEQ    1024
#define DMODEL 512
#define EDIM   128

typedef __attribute__((ext_vector_type(16))) __bf16          v16bf;
typedef __attribute__((ext_vector_type(8)))  float           v8f;
typedef __attribute__((ext_vector_type(8)))  unsigned int    u32x8;
typedef __attribute__((ext_vector_type(4)))  unsigned int    u32x4;
typedef __attribute__((ext_vector_type(4)))  float           f32x4;

// Pack the bf16 truncations (high 16 bits) of two fp32 values into one dword
// with a single v_perm_b32: result = {hi16(h), hi16(l)}.
static __device__ __forceinline__ unsigned int bfpack(float l, float h) {
    return __builtin_amdgcn_perm(__float_as_uint(h), __float_as_uint(l), 0x07060302u);
}

// Build a WMMA bf16 A/B fragment (v16bf) from 16 fp32 values by truncation.
// Elements 0..7 -> VGPRs 0..3 (low K group), 8..15 -> VGPRs 4..7 (high K
// group), matching the CDNA5 16-bit A/B VGPR layout. 8 v_perm_b32 total.
static __device__ __forceinline__ v16bf cvt4(f32x4 a0, f32x4 a1, f32x4 a2, f32x4 a3) {
    u32x8 t;
    t[0] = bfpack(a0[0], a0[1]);
    t[1] = bfpack(a0[2], a0[3]);
    t[2] = bfpack(a1[0], a1[1]);
    t[3] = bfpack(a1[2], a1[3]);
    t[4] = bfpack(a2[0], a2[1]);
    t[5] = bfpack(a2[2], a2[3]);
    t[6] = bfpack(a3[0], a3[1]);
    t[7] = bfpack(a3[2], a3[3]);
    return __builtin_bit_cast(v16bf, t);
}

// Combine two 16-byte bf16 groups (low-K group, high-K group) into a v16bf.
static __device__ __forceinline__ v16bf pack2(u32x4 lo, u32x4 hi) {
    u32x8 t;
#pragma unroll
    for (int i = 0; i < 4; ++i) { t[i] = lo[i]; t[4 + i] = hi[i]; }
    return __builtin_bit_cast(v16bf, t);
}

static __device__ __forceinline__ v16bf ldfrag(const float* row, int k0, int koff) {
    f32x4 a0 = *(const f32x4*)(row + k0 + koff);
    f32x4 a1 = *(const f32x4*)(row + k0 + koff + 4);
    f32x4 a2 = *(const f32x4*)(row + k0 + 16 + koff);
    f32x4 a3 = *(const f32x4*)(row + k0 + 16 + koff + 4);
    return cvt4(a0, a1, a2, a3);
}

// ---------------------------------------------------------------------------
// Kernel 1: Q/K projection GEMM.
// Rows = L*B*N = 98304 (flat), cols = 256 (0..127 -> Q, 128..255 -> K).
// One wave computes a PAIR of adjacent 16x16 output tiles (shared A fragment)
// so two independent v_wmma_f32_16x16x32_bf16 chains interleave and fill each
// other's WMMA data-hazard slots.
// ---------------------------------------------------------------------------
__global__ __launch_bounds__(256)
void proj_kernel(const float* __restrict__ hs,
                 const float* __restrict__ qw, const float* __restrict__ qb,
                 const float* __restrict__ kw, const float* __restrict__ kb,
                 unsigned short* __restrict__ qout,
                 unsigned short* __restrict__ kout) {
    const int wave = blockIdx.x * 8 + (threadIdx.x >> 5);
    const int lane = threadIdx.x & 31;
    const int rt   = wave >> 3;        // row tile (16 rows)
    const int cp   = wave & 7;         // col-tile pair
    const int row0 = rt << 4;
    const int l    = row0 >> 13;       // / (B*N)
    const bool isK = (cp >= 4);        // pairs 0-3 -> Q, 4-7 -> K
    const int e00  = (cp & 3) << 5;    // first tile col base within matrix
    const int e01  = e00 + 16;
    const int col  = lane & 15;
    const int koff = (lane >> 4) << 3; // 0 for lanes 0-15, 8 for lanes 16-31

    const float* W    = (isK ? kw : qw) + (size_t)l * (EDIM * DMODEL);
    const float* bias = (isK ? kb : qb) + l * EDIM;
    unsigned short* out = isK ? kout : qout;

    const float* arow  = hs + (size_t)(row0 + col) * DMODEL;
    const float* brow0 = W  + (size_t)(e00 + col) * DMODEL;
    const float* brow1 = W  + (size_t)(e01 + col) * DMODEL;

    v8f acc0 = {};
    v8f acc1 = {};
#pragma unroll 4
    for (int t = 0; t < 16; ++t) {
        const int k0 = t * 32;
        v16bf a  = ldfrag(arow,  k0, koff);
        v16bf b0 = ldfrag(brow0, k0, koff);
        v16bf b1 = ldfrag(brow1, k0, koff);
        acc0 = __builtin_amdgcn_wmma_f32_16x16x32_bf16(false, a, false, b0,
                                                       (short)0, acc0, false, false);
        acc1 = __builtin_amdgcn_wmma_f32_16x16x32_bf16(false, a, false, b1,
                                                       (short)0, acc1, false, false);
    }

    const float bv0 = bias[e00 + col];
    const float bv1 = bias[e01 + col];
    const size_t ob0 = (size_t)row0 * EDIM + e00 + col;
    const size_t ob1 = (size_t)row0 * EDIM + e01 + col;
#pragma unroll
    for (int r = 0; r < 8; ++r) {
        const int m = r + ((lane >> 4) << 3);
        const float v0 = acc0[r] + bv0;
        const float v1 = acc1[r] + bv1;
        out[ob0 + (size_t)m * EDIM] = (unsigned short)(__float_as_uint(v0) >> 16);
        out[ob1 + (size_t)m * EDIM] = (unsigned short)(__float_as_uint(v1) >> 16);
    }
}

// ---------------------------------------------------------------------------
// Kernel 2: ordering scores s[l,b,n] = hidden . ord_w + ord_b (one wave/row).
// ---------------------------------------------------------------------------
__global__ __launch_bounds__(256)
void score_kernel(const float* __restrict__ hs, const float* __restrict__ ow,
                  const float* __restrict__ ob, float* __restrict__ sout) {
    const int wave = blockIdx.x * 8 + (threadIdx.x >> 5);
    const int lane = threadIdx.x & 31;
    const int l    = wave >> 13;
    const float* row = hs + (size_t)wave * DMODEL;
    const float* w   = ow + l * DMODEL;
    float sum = 0.f;
#pragma unroll
    for (int k = 0; k < 16; ++k) {
        const int idx = lane + k * 32;
        sum = fmaf(row[idx], w[idx], sum);
    }
#pragma unroll
    for (int off = 16; off > 0; off >>= 1) sum += __shfl_xor(sum, off, 32);
    if (lane == 0) sout[wave] = sum + ob[l];
}

// ---------------------------------------------------------------------------
// Kernel 3: graph refinement. One wave per 16x32 output tile of (B,N,N):
// one i-tile x two j-tiles sharing the Q fragment, giving two independent
// WMMA chains per layer. Scan recurrence fused in registers.
// ---------------------------------------------------------------------------
__global__ __launch_bounds__(256)
void graph_kernel(const unsigned short* __restrict__ Q,
                  const unsigned short* __restrict__ K,
                  const float* __restrict__ s,
                  const float* __restrict__ gates,
                  float* __restrict__ out) {
    const int wave = blockIdx.x * 8 + (threadIdx.x >> 5);
    const int lane = threadIdx.x & 31;
    const int b    = wave >> 11;       // / (64*32)
    const int rem  = wave & 2047;
    const int it   = rem >> 5;
    const int jtp  = rem & 31;         // j-tile pair
    const int jt0  = jtp * 2;
    const int jt1  = jt0 + 1;
    const int col  = lane & 15;
    const int hi   = lane >> 4;
    const int koff = hi << 3;          // bf16-element offset (0 or 8)

    const float scale = 0.08838834764831845f; // 128^-0.5

    float acc0[8], acc1[8];
#pragma unroll
    for (int r = 0; r < 8; ++r) {
        const int i  = it * 16 + r + hi * 8;
        const int j0 = jt0 * 16 + col;
        const int j1 = jt1 * 16 + col;
        acc0[r] = (i == j0) ? -100.f : -2.f;  // G_0
        acc1[r] = (i == j1) ? -100.f : -2.f;
    }

    for (int l = 0; l < LAYERS; ++l) {
        const float gate  = 1.f / (1.f + __expf(-gates[l]));
        const float temp  = fmaxf(2.0f * (1.0f - (float)l * (0.9f / 11.0f)), 0.1f);
        const float inv_t = 1.f / temp;
        const float omg   = 1.0f - gate;

        const float* sl = s + (size_t)(l * BATCH + b) * SEQ;
        const float s_col0 = sl[jt0 * 16 + col];
        const float s_col1 = sl[jt1 * 16 + col];

        const unsigned short* qbase =
            Q + ((size_t)(l * BATCH + b) * SEQ + it * 16 + col) * EDIM + koff;
        const unsigned short* kbase0 =
            K + ((size_t)(l * BATCH + b) * SEQ + jt0 * 16 + col) * EDIM + koff;
        const unsigned short* kbase1 =
            K + ((size_t)(l * BATCH + b) * SEQ + jt1 * 16 + col) * EDIM + koff;

        u32x4 qa[8], kb0[8], kb1[8];
#pragma unroll
        for (int g = 0; g < 8; ++g) {
            qa[g]  = *(const u32x4*)(qbase  + g * 16);
            kb0[g] = *(const u32x4*)(kbase0 + g * 16);
            kb1[g] = *(const u32x4*)(kbase1 + g * 16);
        }

        v8f c0 = {};
        v8f c1 = {};
#pragma unroll
        for (int t = 0; t < 4; ++t) {
            v16bf a  = pack2(qa[2 * t], qa[2 * t + 1]);
            v16bf b0 = pack2(kb0[2 * t], kb0[2 * t + 1]);
            v16bf b1 = pack2(kb1[2 * t], kb1[2 * t + 1]);
            c0 = __builtin_amdgcn_wmma_f32_16x16x32_bf16(false, a, false, b0,
                                                         (short)0, c0, false, false);
            c1 = __builtin_amdgcn_wmma_f32_16x16x32_bf16(false, a, false, b1,
                                                         (short)0, c1, false, false);
        }

#pragma unroll
        for (int r = 0; r < 8; ++r) {
            const float s_row = sl[it * 16 + r + hi * 8];
            const float adj0 = scale * c0[r] + 2.0f * tanhf(s_col0 - s_row);
            const float adj1 = scale * c1[r] + 2.0f * tanhf(s_col1 - s_row);
            acc0[r] = (acc0[r] * omg + adj0 * gate) * inv_t;  // scan step
            acc1[r] = (acc1[r] * omg + adj1 * gate) * inv_t;
        }
    }

    const size_t ob0 = ((size_t)b * SEQ + it * 16) * SEQ + jt0 * 16 + col;
    const size_t ob1 = ((size_t)b * SEQ + it * 16) * SEQ + jt1 * 16 + col;
#pragma unroll
    for (int r = 0; r < 8; ++r) {
        out[ob0 + (size_t)(r + hi * 8) * SEQ] = acc0[r];
        out[ob1 + (size_t)(r + hi * 8) * SEQ] = acc1[r];
    }
}

extern "C" void kernel_launch(void* const* d_in, const int* in_sizes, int n_in,
                              void* d_out, int out_size, void* d_ws, size_t ws_size,
                              hipStream_t stream) {
    const float* hs    = (const float*)d_in[0];
    const float* qw    = (const float*)d_in[1];
    const float* qb    = (const float*)d_in[2];
    const float* kw    = (const float*)d_in[3];
    const float* kb    = (const float*)d_in[4];
    const float* ow    = (const float*)d_in[5];
    const float* ob    = (const float*)d_in[6];
    const float* gates = (const float*)d_in[7];
    float* out = (float*)d_out;

    const size_t qk_elems = (size_t)LAYERS * BATCH * SEQ * EDIM; // 12.58M bf16 each
    unsigned short* qws = (unsigned short*)d_ws;
    unsigned short* kws = qws + qk_elems;
    float* sws = (float*)((char*)d_ws + 2 * qk_elems * sizeof(unsigned short));

    // proj: 6144 row-tiles * 8 col-tile-pairs = 49152 waves -> 6144 blocks
    proj_kernel<<<6144, 256, 0, stream>>>(hs, qw, qb, kw, kb, qws, kws);
    // scores: one wave per row, 98304 rows -> 12288 blocks
    score_kernel<<<12288, 256, 0, stream>>>(hs, ow, ob, sws);
    // graph: 8 * 64 * 32 = 16384 waves (16x32 tiles) -> 2048 blocks
    graph_kernel<<<2048, 256, 0, stream>>>(qws, kws, sws, gates, out);
}
